// M_IF_19567871000761
// MI455X (gfx1250) — matrix operand order repaired and checked by reading, not compiled, not run on hardware
//
#include <hip/hip_runtime.h>
#include <hip/hip_bf16.h>

typedef _Float16 h2   __attribute__((ext_vector_type(2)));
typedef __attribute__((ext_vector_type(16))) _Float16 v16h;
typedef __attribute__((ext_vector_type(8)))  float    v8f;

static constexpr int kC   = 128;     // channels
static constexpr int kHW  = 4096;    // 64*64 spatial
static constexpr int kQKV = 384;     // 3*C

__device__ __forceinline__ float bperm(float v, int srcLane) {
    return __int_as_float(__builtin_amdgcn_ds_bpermute(srcLane << 2, __float_as_int(v)));
}

// ---------------------------------------------------------------------------
// Kernel 1: GroupNorm (16 groups of 8 channels). One block per (tensor,group).
// Writes fp32 normalized tensor (residual) and f16 copy (GEMM input).
// ---------------------------------------------------------------------------
__global__ __launch_bounds__(256) void gn_kernel(
    const float* __restrict__ xA, const float* __restrict__ xB,
    const float* __restrict__ w,  const float* __restrict__ b,
    float* __restrict__ xnA, float* __restrict__ xnB,
    _Float16* __restrict__ xnAh, _Float16* __restrict__ xnBh) {
    const int t = blockIdx.x >> 4;       // tensor: 0=A, 1=B
    const int g = blockIdx.x & 15;       // group
    const float*    X  = t ? xB  : xA;
    float*          XN = t ? xnB : xnA;
    _Float16*       XH = t ? xnBh : xnAh;
    const int  c0  = g * 8;
    const int  cnt = 8 * kHW;            // 32768 elements per group

    float s = 0.f, ss = 0.f;
    for (int idx = threadIdx.x; idx < cnt; idx += 256) {
        float v = X[c0 * kHW + idx];
        s += v; ss += v * v;
    }
    __shared__ float red[512];
    red[threadIdx.x] = s; red[256 + threadIdx.x] = ss;
    __syncthreads();
    for (int o = 128; o > 0; o >>= 1) {
        if ((int)threadIdx.x < o) {
            red[threadIdx.x]       += red[threadIdx.x + o];
            red[256 + threadIdx.x] += red[256 + threadIdx.x + o];
        }
        __syncthreads();
    }
    const float mu   = red[0] / (float)cnt;
    const float var  = red[256] / (float)cnt - mu * mu;
    const float rstd = rsqrtf(var + 1e-5f);

    for (int idx = threadIdx.x; idx < cnt; idx += 256) {
        int   ch = c0 + (idx >> 12);            // idx / 4096
        float v  = (X[c0 * kHW + idx] - mu) * rstd * w[ch] + b[ch];
        XN[c0 * kHW + idx] = v;
        XH[c0 * kHW + idx] = (_Float16)v;
    }
}

// ---------------------------------------------------------------------------
// Kernel 2: convert the four weight matrices to f16.
// ---------------------------------------------------------------------------
__global__ __launch_bounds__(256) void cvt_kernel(
    const float* __restrict__ wqA, const float* __restrict__ wqB,
    const float* __restrict__ woA, const float* __restrict__ woB,
    _Float16* __restrict__ hqA, _Float16* __restrict__ hqB,
    _Float16* __restrict__ hoA, _Float16* __restrict__ hoB) {
    int i = blockIdx.x * 256 + threadIdx.x;
    if (i < kQKV * kC) { hqA[i] = (_Float16)wqA[i]; hqB[i] = (_Float16)wqB[i]; }
    if (i < kC * kC)   { hoA[i] = (_Float16)woA[i]; hoB[i] = (_Float16)woB[i]; }
}

// ---------------------------------------------------------------------------
// Kernel 3: QKV GEMM.  qkv[384,4096] = W[384,128] @ Xh[128,4096], f16 WMMA.
// Each wave computes one 16x16 tile (K = 128 -> 4 WMMAs). Q rows scaled by
// 1/sqrt(C) on store.
// ---------------------------------------------------------------------------
__global__ __launch_bounds__(256) void qkv_gemm_kernel(
    const _Float16* __restrict__ WA, const _Float16* __restrict__ WB,
    const _Float16* __restrict__ XA, const _Float16* __restrict__ XB,
    _Float16* __restrict__ QKVA, _Float16* __restrict__ QKVB) {
    const int wave = threadIdx.x >> 5, lane = threadIdx.x & 31;
    const int tile = blockIdx.x * 8 + wave;       // 2 * 24 * 256 tiles total
    const int dir  = tile / 6144;
    const int rem  = tile - dir * 6144;
    const int mt   = rem >> 8;                    // 0..23
    const int nt   = rem & 255;                   // 0..255
    const _Float16* W = dir ? WB : WA;
    const _Float16* X = dir ? XB : XA;
    _Float16*       O = dir ? QKVB : QKVA;

    const int li = lane & 15, hh = lane >> 4;
    v8f acc = {};
    for (int kt = 0; kt < 4; ++kt) {
        v16h a = {}, bf = {};
#pragma unroll
        for (int rr = 0; rr < 8; ++rr) {          // A frag: lane m = li
            int kd = (rr < 4) ? (8 * hh + 2 * rr) : (16 + 8 * hh + 2 * (rr - 4));
            h2 tv = *(const h2*)(W + (mt * 16 + li) * kC + kt * 32 + kd);
            a[2 * rr] = tv.x; a[2 * rr + 1] = tv.y;
        }
        // B frag: lane n = li
#pragma unroll
        for (int v = 0; v < 8; ++v) {
            int kk = (v < 4) ? (8 * hh + 2 * v) : (16 + 8 * hh + 2 * (v - 4));
            bf[2 * v]     = X[(kt * 32 + kk) * kHW + nt * 16 + li];
            bf[2 * v + 1] = X[(kt * 32 + kk + 1) * kHW + nt * 16 + li];
        }
        acc = __builtin_amdgcn_wmma_f32_16x16x32_f16(false, a, false, bf,
                                                     (short)0, acc, false, false);
    }
    const float qscale = 0.08838834764831845f;    // 1/sqrt(128)
#pragma unroll
    for (int rr = 0; rr < 8; ++rr) {
        int   row = mt * 16 + rr + 8 * hh;
        float vv  = acc[rr];
        if (row < kC) vv *= qscale;               // scale only Q rows
        O[row * kHW + nt * 16 + li] = (_Float16)vv;
    }
}

// ---------------------------------------------------------------------------
// Kernel 4: flash attention.  dir0: O_A = attend(qB, kA, vA); dir1 swaps.
// Block = 8 waves, each wave owns a 16-query tile (128 queries / block).
// K,V chunks (32 x 64 f16 each) are staged in LDS in global [d][j] order via
// GLOBAL_LOAD_ASYNC_TO_LDS_B128 (ASYNCcnt-tracked), double-buffered so chunk
// c+1 streams in while chunk c is consumed by the WMMA pipeline.
// S^T tiles via WMMA (C-layout lane = query i) -> per-lane online softmax;
// P A-fragment is a pure in-lane pack of two S^T tiles (no cross-lane
// transpose); O accumulated as O^T[i,d] in two 16-wide d-chunks.
// ---------------------------------------------------------------------------
__global__ __launch_bounds__(256) void attn_kernel(
    const _Float16* __restrict__ QKVA, const _Float16* __restrict__ QKVB,
    _Float16* __restrict__ OA, _Float16* __restrict__ OB) {
    // [buf][d][j] with row padding 72 (144B stride, 16B-aligned rows)
    __shared__ __align__(16) _Float16 Kl[2][32][72];
    __shared__ __align__(16) _Float16 Vl[2][32][72];

    const int dir  = blockIdx.x >> 7;
    const int rest = blockIdx.x & 127;
    const int head = rest >> 5;
    const int iblk = rest & 31;
    const int wave = threadIdx.x >> 5, lane = threadIdx.x & 31;
    const int i0   = iblk * 128 + wave * 16;

    const _Float16* Q = (dir ? QKVA : QKVB) + (head * 32) * kHW;          // queries
    const _Float16* K = (dir ? QKVB : QKVA) + (kC + head * 32) * kHW;     // keys
    const _Float16* V = (dir ? QKVB : QKVA) + (2 * kC + head * 32) * kHW; // values
    _Float16*       O = dir ? OB : OA;

    const int li = lane & 15, hh = lane >> 4;

    // staging coords: thread -> (d row, 8-wide j sub-chunk), 16B per thread
    const int sd  = threadIdx.x >> 3;        // 0..31
    const int sj8 = (threadIdx.x & 7) * 8;   // 0,8,...,56

    // Q B-fragment (persistent): lane n = i = li
    v16h qb = {};
#pragma unroll
    for (int v = 0; v < 8; ++v) {
        int dq = (v < 4) ? (8 * hh + 2 * v) : (16 + 8 * hh + 2 * (v - 4));
        qb[2 * v]     = Q[dq * kHW + i0 + li];
        qb[2 * v + 1] = Q[(dq + 1) * kHW + i0 + li];
    }

    float mrow = -3.0e38f, lrow = 0.f;
    v8f o0 = {}, o1 = {};

    auto issue_chunk = [&](int c, int buf) {
        const _Float16* gk = K + sd * kHW + c * 64 + sj8;
        const _Float16* gv = V + sd * kHW + c * 64 + sj8;
        unsigned lk = (unsigned)(size_t)&Kl[buf][sd][sj8];  // LDS byte offset
        unsigned lv = (unsigned)(size_t)&Vl[buf][sd][sj8];
        asm volatile("global_load_async_to_lds_b128 %0, %1, off"
                     :: "v"(lk), "v"(gk) : "memory");
        asm volatile("global_load_async_to_lds_b128 %0, %1, off"
                     :: "v"(lv), "v"(gv) : "memory");
    };

    const int NC = kHW / 64;                 // 64 chunks of 64 keys
    issue_chunk(0, 0);

    for (int c = 0; c < NC; ++c) {
        if (c + 1 < NC) {
            issue_chunk(c + 1, (c + 1) & 1);
            // 2 newest async loads (chunk c+1) may remain outstanding;
            // loads complete in order so chunk c has landed.
            asm volatile("s_wait_asynccnt 2" ::: "memory");
        } else {
            asm volatile("s_wait_asynccnt 0" ::: "memory");
        }
        __syncthreads();
        const int buf = c & 1;

        for (int js = 0; js < 64; js += 32) {
            // Two S^T tiles: A = K^T (16j x 32d), B = Q (32d x 16i)
            v16h ka0 = {}, ka1 = {};
#pragma unroll
            for (int rr = 0; rr < 8; ++rr) {
                int kd = (rr < 4) ? (8 * hh + 2 * rr) : (16 + 8 * hh + 2 * (rr - 4));
                ka0[2 * rr]     = Kl[buf][kd][js + li];
                ka0[2 * rr + 1] = Kl[buf][kd + 1][js + li];
                ka1[2 * rr]     = Kl[buf][kd][js + 16 + li];
                ka1[2 * rr + 1] = Kl[buf][kd + 1][js + 16 + li];
            }
            v8f z = {};
            v8f st0 = __builtin_amdgcn_wmma_f32_16x16x32_f16(false, ka0, false, qb,
                                                             (short)0, z, false, false);
            v8f st1 = __builtin_amdgcn_wmma_f32_16x16x32_f16(false, ka1, false, qb,
                                                             (short)0, z, false, false);
            // online softmax (lane = query i, halves combined via xor-16 bpermute)
            float tmax = -3.0e38f;
#pragma unroll
            for (int rr = 0; rr < 8; ++rr)
                tmax = fmaxf(tmax, fmaxf(st0[rr], st1[rr]));
            tmax = fmaxf(tmax, bperm(tmax, lane ^ 16));
            float mn  = fmaxf(mrow, tmax);
            float cor = __expf(mrow - mn);
            float rs  = 0.f;
            v16h pa = {};                         // P A-frag: pure in-lane pack
#pragma unroll
            for (int rr = 0; rr < 8; ++rr) {
                float p0 = __expf(st0[rr] - mn);
                float p1 = __expf(st1[rr] - mn);
                rs += p0 + p1;
                pa[rr]     = (_Float16)p0;
                pa[8 + rr] = (_Float16)p1;
            }
            rs  += bperm(rs, lane ^ 16);
            lrow = lrow * cor + rs;
            mrow = mn;
            // rescale O accumulators: row i = rr + 8*hh -> c_i from lane rr+8*hh
#pragma unroll
            for (int rr = 0; rr < 8; ++rr) {
                float cb = bperm(cor, rr + 8 * hh);
                o0[rr] *= cb; o1[rr] *= cb;
            }
            // V^T B-frags (32j x 16d) for d = 0..15 and 16..31; lane n = d
            v16h vb0 = {}, vb1 = {};
#pragma unroll
            for (int v = 0; v < 8; ++v) {
                int jb = js + ((v < 4) ? (8 * hh + 2 * v) : (16 + 8 * hh + 2 * (v - 4)));
                vb0[2 * v]     = Vl[buf][li][jb];
                vb0[2 * v + 1] = Vl[buf][li][jb + 1];
                vb1[2 * v]     = Vl[buf][16 + li][jb];
                vb1[2 * v + 1] = Vl[buf][16 + li][jb + 1];
            }
            o0 = __builtin_amdgcn_wmma_f32_16x16x32_f16(false, pa, false, vb0,
                                                        (short)0, o0, false, false);
            o1 = __builtin_amdgcn_wmma_f32_16x16x32_f16(false, pa, false, vb1,
                                                        (short)0, o1, false, false);
        }
        __syncthreads();   // all reads of buf done -> next iter may refill it
    }
    // finalize: O[d, i] = O^T[i, d] / l_i
    const float linv = 1.f / lrow;
#pragma unroll
    for (int rr = 0; rr < 8; ++rr) {
        float wsc = bperm(linv, rr + 8 * hh);
        int   col = i0 + rr + 8 * hh;
        O[(head * 32 + li) * kHW + col]      = (_Float16)(o0[rr] * wsc);
        O[(head * 32 + 16 + li) * kHW + col] = (_Float16)(o1[rr] * wsc);
    }
}

// ---------------------------------------------------------------------------
// Kernel 5: output GEMM + bias + normalized residual -> d_out (fp32).
// out[128,4096] = Wout[128,128] @ O[128,4096] + bias + xn
// ---------------------------------------------------------------------------
__global__ __launch_bounds__(256) void out_gemm_kernel(
    const _Float16* __restrict__ WhA, const _Float16* __restrict__ WhB,
    const _Float16* __restrict__ OhA, const _Float16* __restrict__ OhB,
    const float* __restrict__ bA, const float* __restrict__ bB,
    const float* __restrict__ xnA, const float* __restrict__ xnB,
    float* __restrict__ out) {
    const int wave = threadIdx.x >> 5, lane = threadIdx.x & 31;
    const int tile = blockIdx.x * 8 + wave;       // 2 * 8 * 256 tiles
    const int dir  = tile / 2048;
    const int rem  = tile - dir * 2048;
    const int mt   = rem >> 8;                    // 0..7
    const int nt   = rem & 255;                   // 0..255
    const _Float16* W    = dir ? WhB : WhA;
    const _Float16* Bm   = dir ? OhB : OhA;
    const float*    bias = dir ? bB : bA;
    const float*    xn   = dir ? xnB : xnA;
    float*          outp = out + dir * kC * kHW;

    const int li = lane & 15, hh = lane >> 4;
    v8f acc = {};
    for (int kt = 0; kt < 4; ++kt) {
        v16h a = {}, bf = {};
#pragma unroll
        for (int rr = 0; rr < 8; ++rr) {
            int kd = (rr < 4) ? (8 * hh + 2 * rr) : (16 + 8 * hh + 2 * (rr - 4));
            h2 tv = *(const h2*)(W + (mt * 16 + li) * kC + kt * 32 + kd);
            a[2 * rr] = tv.x; a[2 * rr + 1] = tv.y;
        }
#pragma unroll
        for (int v = 0; v < 8; ++v) {
            int kk = (v < 4) ? (8 * hh + 2 * v) : (16 + 8 * hh + 2 * (v - 4));
            bf[2 * v]     = Bm[(kt * 32 + kk) * kHW + nt * 16 + li];
            bf[2 * v + 1] = Bm[(kt * 32 + kk + 1) * kHW + nt * 16 + li];
        }
        acc = __builtin_amdgcn_wmma_f32_16x16x32_f16(false, a, false, bf,
                                                     (short)0, acc, false, false);
    }
#pragma unroll
    for (int rr = 0; rr < 8; ++rr) {
        int row = mt * 16 + rr + 8 * hh;
        int col = nt * 16 + li;
        outp[row * kHW + col] = acc[rr] + bias[row] + xn[row * kHW + col];
    }
}

// ---------------------------------------------------------------------------
extern "C" void kernel_launch(void* const* d_in, const int* in_sizes, int n_in,
                              void* d_out, int out_size, void* d_ws, size_t ws_size,
                              hipStream_t stream) {
    const float* x_A    = (const float*)d_in[0];
    const float* x_B    = (const float*)d_in[1];
    const float* gnA_w  = (const float*)d_in[2];
    const float* gnA_b  = (const float*)d_in[3];
    const float* qkvA_w = (const float*)d_in[4];
    const float* outA_w = (const float*)d_in[5];
    const float* outA_b = (const float*)d_in[6];
    const float* qkvB_w = (const float*)d_in[7];
    const float* outB_w = (const float*)d_in[8];
    const float* outB_b = (const float*)d_in[9];

    char* ws = (char*)d_ws;
    float*    xnA    = (float*)(ws + 0);                    //  2 MB
    float*    xnB    = (float*)(ws + 2097152);              //  2 MB
    _Float16* xnAh   = (_Float16*)(ws + 4194304);           //  1 MB
    _Float16* xnBh   = (_Float16*)(ws + 5242880);           //  1 MB
    _Float16* qkvAh  = (_Float16*)(ws + 6291456);           //  3 MB
    _Float16* qkvBh  = (_Float16*)(ws + 9437184);           //  3 MB
    _Float16* oAh    = (_Float16*)(ws + 12582912);          //  1 MB
    _Float16* oBh    = (_Float16*)(ws + 13631488);          //  1 MB
    _Float16* wqAh   = (_Float16*)(ws + 14680064);
    _Float16* wqBh   = (_Float16*)(ws + 14778368);
    _Float16* woAh   = (_Float16*)(ws + 14876672);
    _Float16* woBh   = (_Float16*)(ws + 14909440);

    gn_kernel<<<32, 256, 0, stream>>>(x_A, x_B, gnA_w, gnA_b, xnA, xnB, xnAh, xnBh);
    cvt_kernel<<<192, 256, 0, stream>>>(qkvA_w, qkvB_w, outA_w, outB_w,
                                        wqAh, wqBh, woAh, woBh);
    qkv_gemm_kernel<<<1536, 256, 0, stream>>>(wqAh, wqBh, xnAh, xnBh, qkvAh, qkvBh);
    attn_kernel<<<256, 256, 0, stream>>>(qkvAh, qkvBh, oAh, oBh);
    out_gemm_kernel<<<512, 256, 0, stream>>>(woAh, woBh, oAh, oBh,
                                             outA_b, outB_b, xnA, xnB,
                                             (float*)d_out);
}
